// AqlmOFTLinear_12635793785537
// MI455X (gfx1250) — matrix operand order repaired and checked by reading, not compiled, not run on hardware
//
#include <hip/hip_runtime.h>
#include <cstdint>
#include <cstddef>

// ---------------------------------------------------------------------------
// AQLM + OFT linear for MI455X (gfx1250, wave32).
//  K1 cayley   : 128x Gauss-Jordan on [(I+S)|(I-S)]  -> Y = Q^T   (fp32)
//  K2 dequant  : codebook gather -> W bf16 hi/lo, natural [o][d] layout
//  K3 rotate   : xr = xb @ Q per 32-block via v_wmma_f32_16x16x4_f32
//                (exact fp32 on the matrix engine), split to bf16 hi/lo
//  K4 gemm     : out = xr @ W^T + bias via v_wmma_f32_16x16x32_bf16,
//                bf16x3 split (hi*hi + hi*lo + lo*hi) ~ fp32 accuracy,
//                double-buffered LDS, async global->LDS staging (ASYNCcnt).
// Workspace requirement: ~321 MB (1MB Q + 256MB xr hi/lo + 64MB W hi/lo).
// ---------------------------------------------------------------------------

typedef __bf16 bf16_t;
typedef __attribute__((ext_vector_type(16))) __bf16 v16bf;
typedef __attribute__((ext_vector_type(8)))  __bf16 v8bf;
typedef __attribute__((ext_vector_type(8)))  float  v8f;
typedef __attribute__((ext_vector_type(4)))  float  v4f;
typedef __attribute__((ext_vector_type(2)))  float  v2f;
typedef __attribute__((ext_vector_type(4)))  int    v4i;

// exact pointer types for the async builtin (param0: v4i in AS1; dst: AS3)
typedef __attribute__((address_space(1))) v4i* as1_v4i_p;
typedef __attribute__((address_space(3))) v4i* as3_v4i_p;

constexpr int IN_F   = 4096;
constexpr int OUT_F  = 4096;
constexpr int BLOCK  = 32;
constexpr int NBLK   = IN_F / BLOCK;   // 128
constexpr int GROUP  = 8;
constexpr int NGRP   = IN_F / GROUP;   // 512
constexpr int TOKENS = 4 * 4096;       // 16384

// --- gfx1250 async global->LDS staging (confirmed present on this toolchain,
// keep the hedge anyway) ---
#if defined(__has_builtin)
#  if __has_builtin(__builtin_amdgcn_global_load_async_to_lds_b128) && \
      __has_builtin(__builtin_amdgcn_s_wait_asynccnt)
#    define USE_ASYNC_LDS 1
#  endif
#  if __has_builtin(__builtin_amdgcn_wmma_f32_16x16x4_f32)
#    define USE_WMMA_F32 1
#  endif
#endif
#ifndef USE_ASYNC_LDS
#  define USE_ASYNC_LDS 0
#endif
#ifndef USE_WMMA_F32
#  define USE_WMMA_F32 0
#endif

static __device__ __forceinline__ void cp16_g2l(bf16_t* dst, const bf16_t* src) {
#if USE_ASYNC_LDS
  __builtin_amdgcn_global_load_async_to_lds_b128(
      (as1_v4i_p)(uintptr_t)(src),
      (as3_v4i_p)(uintptr_t)(dst),
      0, 0);
#else
  *reinterpret_cast<v8bf*>(dst) = *reinterpret_cast<const v8bf*>(src);
#endif
}

static __device__ __forceinline__ void wait_stage() {
#if USE_ASYNC_LDS
  __builtin_amdgcn_s_wait_asynccnt(0);
#endif
}

union FragBF {
  v16bf v;
  v8bf  h[2];
};

static __device__ __forceinline__ v8f wmma_bf16(v16bf a, v16bf b, v8f c) {
  return __builtin_amdgcn_wmma_f32_16x16x32_bf16(false, a, false, b,
                                                 (short)0, c, false, false);
}

static __device__ __forceinline__ void split_f32(float x, bf16_t& hi, bf16_t& lo) {
  hi = (bf16_t)x;                 // RNE to bf16
  lo = (bf16_t)(x - (float)hi);   // residual, also bf16
}

// ---------------------------------------------------------------------------
// K1: Cayley via Gauss-Jordan. S skew-symmetric => solve (I+S) Y = (I-S);
// Y = Q^T stored row-major (row k = column k of Q).
// ---------------------------------------------------------------------------
__global__ void cayley_kernel(const float* __restrict__ R, float* __restrict__ Qt) {
  __shared__ float aug[32][65];
  __shared__ float rowk[64];
  __shared__ float fcol[32];
  const int n   = blockIdx.x;
  const int tid = threadIdx.x;
  const float* Rb = R + (size_t)n * BLOCK * BLOCK;

  for (int idx = tid; idx < 32 * 64; idx += 256) {
    int i = idx >> 6, j = idx & 63, jj = j & 31;
    float s = 0.5f * (Rb[i * 32 + jj] - Rb[jj * 32 + i]);
    float d = (i == jj) ? 1.0f : 0.0f;
    aug[i][j] = (j < 32) ? (d + s) : (d - s);
  }
  __syncthreads();

  for (int k = 0; k < 32; ++k) {
    if (tid < 64) {
      rowk[tid] = aug[k][tid] / aug[k][k];   // diagonally dominant: no pivoting
    } else if (tid < 96) {
      fcol[tid - 64] = aug[tid - 64][k];
    }
    __syncthreads();
    for (int idx = tid; idx < 32 * 64; idx += 256) {
      int i = idx >> 6, j = idx & 63;
      if (i == k) aug[i][j] = rowk[j];
      else        aug[i][j] -= fcol[i] * rowk[j];
    }
    __syncthreads();
  }

  for (int idx = tid; idx < 32 * 32; idx += 256) {
    int i = idx >> 5, j = idx & 31;
    Qt[(size_t)n * 1024 + idx] = aug[i][32 + j];
  }
}

// ---------------------------------------------------------------------------
// K2: dequantize W, bf16 hi/lo, NATURAL [o][d] layout (16B store per group).
// ---------------------------------------------------------------------------
__global__ void dequant_kernel(const int*   __restrict__ codes,
                               const float* __restrict__ codebooks,
                               const float* __restrict__ scales,
                               bf16_t* __restrict__ wh, bf16_t* __restrict__ wl) {
  const int o  = blockIdx.x * 256 + threadIdx.x;
  const int g0 = blockIdx.y * (NGRP / 8);
  const float s = scales[o];
  for (int g = g0; g < g0 + NGRP / 8; ++g) {
    const int idx = codes[(size_t)o * NGRP + g];
    const v4f* cb = reinterpret_cast<const v4f*>(codebooks + (size_t)idx * GROUP);
    v4f c0 = cb[0], c1 = cb[1];
    v8bf hv, lv;
#pragma unroll
    for (int i = 0; i < 8; ++i) {
      float v = ((i < 4) ? c0[i] : c1[i - 4]) * s;
      bf16_t hi, lo; split_f32(v, hi, lo);
      hv[i] = hi; lv[i] = lo;
    }
    size_t d = (size_t)o * IN_F + (size_t)g * GROUP;
    *reinterpret_cast<v8bf*>(wh + d) = hv;
    *reinterpret_cast<v8bf*>(wl + d) = lv;
  }
}

// ---------------------------------------------------------------------------
// K3: per-block rotation, exact fp32 on the matrix engine.
// WG = (block n, 128 tokens); wave w owns tokens [w*16, w*16+16), computes a
// 16x32 tile as 2 x (16x16) accumulators over K=32 in 8 steps of K=4 using
// v_wmma_f32_16x16x4_f32. Q stored un-transposed in LDS (Qs[j][k]) so
// B-fragment loads are lane-consecutive.
// ---------------------------------------------------------------------------
__global__ void __launch_bounds__(256) rotate_kernel(
    const float* __restrict__ x, const float* __restrict__ Qt,
    bf16_t* __restrict__ xrh, bf16_t* __restrict__ xrl) {
  __shared__ __align__(16) float xs[128][36];  // 128 tokens x 32 feats, pad 36
  __shared__ float Qs[32][33];                 // Qs[j][k] = Q[j][k] = Y[k][j]
  const int n   = blockIdx.x;
  const int t0  = blockIdx.y * 128;
  const int tid = threadIdx.x;

  for (int idx = tid; idx < 1024; idx += 256) {
    int k = idx >> 5, j = idx & 31;            // Qt holds Y[k][j]
    Qs[j][k] = Qt[(size_t)n * 1024 + idx];
  }
  for (int idx = tid; idx < 1024; idx += 256) {
    int r = idx >> 3, c4 = (idx & 7) * 4;      // 128 rows x 8 v4f chunks
    *reinterpret_cast<v4f*>(&xs[r][c4]) =
        *reinterpret_cast<const v4f*>(x + (size_t)(t0 + r) * IN_F + n * 32 + c4);
  }
  __syncthreads();

  const int lane = tid & 31;
  const int wave = tid >> 5;
  const int m0w  = wave * 16;
  const int col  = lane & 15;
  const int koff = (lane < 16) ? 0 : 2;        // A: K{0,1} vs K{2,3}; B same

#if USE_WMMA_F32
  v8f acc[2];
  const v8f vzero = {0, 0, 0, 0, 0, 0, 0, 0};
  acc[0] = vzero; acc[1] = vzero;
  const int m = m0w + col;                     // A: M = lane%16
#pragma unroll
  for (int ks = 0; ks < 8; ++ks) {
    const int k0 = ks * 4;
    v2f a = *reinterpret_cast<const v2f*>(&xs[m][k0 + koff]);   // K, K+1
#pragma unroll
    for (int ni = 0; ni < 2; ++ni) {
      v2f b;
      b[0] = Qs[k0 + koff + 0][ni * 16 + col];
      b[1] = Qs[k0 + koff + 1][ni * 16 + col];
      acc[ni] = __builtin_amdgcn_wmma_f32_16x16x4_f32(false, a, false, b,
                                                      (short)0, acc[ni],
                                                      false, false);
    }
  }
  // D layout: VGPR r -> (M = r + 8*(lane>=16), N = lane&15)
  const int mh = (lane < 16) ? 0 : 8;
#pragma unroll
  for (int ni = 0; ni < 2; ++ni) {
#pragma unroll
    for (int r = 0; r < 8; ++r) {
      int row = t0 + m0w + r + mh;
      int c   = n * 32 + ni * 16 + col;
      bf16_t hi, lo; split_f32(acc[ni][r], hi, lo);
      size_t d = (size_t)row * IN_F + c;
      xrh[d] = hi; xrl[d] = lo;
    }
  }
#else
  // VALU fallback: thread (wave, lane) -> output column k, 16 tokens
  const int k = lane;
  for (int r = 0; r < 16; ++r) {
    float acc = 0.0f;
#pragma unroll
    for (int j = 0; j < 32; ++j) acc = fmaf(xs[m0w + r][j], Qs[j][k], acc);
    bf16_t hi, lo; split_f32(acc, hi, lo);
    size_t d = (size_t)(t0 + m0w + r) * IN_F + n * 32 + k;
    xrh[d] = hi; xrl[d] = lo;
  }
#endif
}

// ---------------------------------------------------------------------------
// K4: out(16384x4096) = xr @ W^T + bias. bf16x3 split WMMA GEMM.
// Block: 128x128 tile, 8 waves, wave = 32x64 = 2x4 tiles of 16x16.
// K-step 32; double-buffered LDS (80KB), async 16B global->LDS staging.
// ---------------------------------------------------------------------------
__global__ void __launch_bounds__(256) gemm_kernel(
    const bf16_t* __restrict__ xrh, const bf16_t* __restrict__ xrl,
    const bf16_t* __restrict__ wh,  const bf16_t* __restrict__ wl,
    const float*  __restrict__ bias, float* __restrict__ out) {
  // smem[buf][mat][row][col]; mat: 0=A_hi 1=A_lo 2=B_hi 3=B_lo
  __shared__ __align__(16) bf16_t smem[2][4][128][40];

  const int n0   = blockIdx.x * 128;
  const int m0   = blockIdx.y * 128;
  const int tid  = threadIdx.x;
  const int lane = tid & 31;
  const int wave = tid >> 5;
  const int m_wave = (wave & 3) * 32;   // 4 waves along M
  const int n_wave = (wave >> 2) * 64;  // 2 waves along N

  v8f acc[2][4];
  const v8f vzero = {0, 0, 0, 0, 0, 0, 0, 0};
#pragma unroll
  for (int mi = 0; mi < 2; ++mi)
#pragma unroll
    for (int ni = 0; ni < 4; ++ni) acc[mi][ni] = vzero;

  auto stage = [&](int buf, int k0) {
#pragma unroll
    for (int ci = 0; ci < 2; ++ci) {
      int idx = tid + ci * 256;       // 512 chunks of 8 bf16
      int r   = idx >> 2;             // 0..127
      int c   = (idx & 3) * 8;        // 0,8,16,24
      size_t ga = (size_t)(m0 + r) * IN_F + k0 + c;
      size_t gb = (size_t)(n0 + r) * IN_F + k0 + c;
      cp16_g2l(&smem[buf][0][r][c], xrh + ga);
      cp16_g2l(&smem[buf][1][r][c], xrl + ga);
      cp16_g2l(&smem[buf][2][r][c], wh + gb);
      cp16_g2l(&smem[buf][3][r][c], wl + gb);
    }
  };

  stage(0, 0);

  for (int kt = 0; kt < IN_F / 32; ++kt) {
    wait_stage();
    __syncthreads();
    const int buf = kt & 1;
    if (kt + 1 < IN_F / 32) stage(buf ^ 1, (kt + 1) * 32);

    FragBF ah[2], al[2], bh[4], bl[4];
    const int c0a = (lane < 16) ? 0 : 16;  // A: K 0-15 vs 16-31
    const int c0b = (lane < 16) ? 0 : 16;  // B: K 0-15 vs 16-31, col = lane&15
#pragma unroll
    for (int mi = 0; mi < 2; ++mi) {
      int m = m_wave + mi * 16 + (lane & 15);
      ah[mi].h[0] = *reinterpret_cast<const v8bf*>(&smem[buf][0][m][c0a]);
      ah[mi].h[1] = *reinterpret_cast<const v8bf*>(&smem[buf][0][m][c0a + 8]);
      al[mi].h[0] = *reinterpret_cast<const v8bf*>(&smem[buf][1][m][c0a]);
      al[mi].h[1] = *reinterpret_cast<const v8bf*>(&smem[buf][1][m][c0a + 8]);
    }
#pragma unroll
    for (int ni = 0; ni < 4; ++ni) {
      int nn = n_wave + ni * 16 + (lane & 15);
      bh[ni].h[0] = *reinterpret_cast<const v8bf*>(&smem[buf][2][nn][c0b]);
      bh[ni].h[1] = *reinterpret_cast<const v8bf*>(&smem[buf][2][nn][c0b + 8]);
      bl[ni].h[0] = *reinterpret_cast<const v8bf*>(&smem[buf][3][nn][c0b]);
      bl[ni].h[1] = *reinterpret_cast<const v8bf*>(&smem[buf][3][nn][c0b + 8]);
    }

#pragma unroll
    for (int mi = 0; mi < 2; ++mi)
#pragma unroll
      for (int ni = 0; ni < 4; ++ni) {
        acc[mi][ni] = wmma_bf16(ah[mi].v, bh[ni].v, acc[mi][ni]);
        acc[mi][ni] = wmma_bf16(ah[mi].v, bl[ni].v, acc[mi][ni]);
        acc[mi][ni] = wmma_bf16(al[mi].v, bh[ni].v, acc[mi][ni]);
      }
  }

  const int nl = lane & 15;
  const int mh = (lane < 16) ? 0 : 8;
#pragma unroll
  for (int ni = 0; ni < 4; ++ni) {
    int col  = n0 + n_wave + ni * 16 + nl;
    float bv = bias[col];
#pragma unroll
    for (int mi = 0; mi < 2; ++mi) {
#pragma unroll
      for (int r = 0; r < 8; ++r) {
        int row = m0 + m_wave + mi * 16 + r + mh;
        out[(size_t)row * OUT_F + col] = acc[mi][ni][r] + bv;
      }
    }
  }
}

// ---------------------------------------------------------------------------
extern "C" void kernel_launch(void* const* d_in, const int* in_sizes, int n_in,
                              void* d_out, int out_size, void* d_ws, size_t ws_size,
                              hipStream_t stream) {
  (void)in_sizes; (void)n_in; (void)out_size; (void)ws_size;
  const float* x         = (const float*)d_in[0];
  const float* oft_r     = (const float*)d_in[1];
  const int*   codes     = (const int*)d_in[2];
  const float* codebooks = (const float*)d_in[3];
  const float* scales    = (const float*)d_in[4];
  const float* bias      = (const float*)d_in[5];
  float* out = (float*)d_out;

  char*   wsb = (char*)d_ws;
  float*  Qt  = (float*)wsb;                       // 512 KB
  bf16_t* xrh = (bf16_t*)(wsb + (size_t)(1 << 20));
  bf16_t* xrl = xrh + (size_t)TOKENS * IN_F;       // 128 MB each
  bf16_t* whp = xrl + (size_t)TOKENS * IN_F;
  bf16_t* wlp = whp + (size_t)OUT_F * IN_F;        // 32 MB each

  cayley_kernel <<<dim3(NBLK),               dim3(256), 0, stream>>>(oft_r, Qt);
  dequant_kernel<<<dim3(OUT_F / 256, 8),     dim3(256), 0, stream>>>(codes, codebooks, scales, whp, wlp);
  rotate_kernel <<<dim3(NBLK, TOKENS / 128), dim3(256), 0, stream>>>(x, Qt, xrh, xrl);
  gemm_kernel   <<<dim3(OUT_F / 128, TOKENS / 128), dim3(256), 0, stream>>>(xrh, xrl, whp, wlp, bias, out);
}
